// BasicAttentionBlock_28475633172932
// MI455X (gfx1250) — matrix operand order repaired and checked by reading, not compiled, not run on hardware
//
#include <hip/hip_runtime.h>
#include <hip/hip_bf16.h>

#define INSZ  256   // input_size
#define HSZ   128   // hidden_size
#define OUTSZ 128   // NH*HD
#define NHEAD 8
#define HDIM  16

#define WI_PAD (INSZ + 16)   // bf16 row pitch for staged w_in  (544 B, 32B-aligned rows)
#define WH_PAD (HSZ + 16)    // bf16 row pitch for staged w_proj/w_skip/h (288 B)

typedef __attribute__((ext_vector_type(16))) __bf16 v16bf;
typedef __attribute__((ext_vector_type(8)))  __bf16 v8bf;
typedef __attribute__((ext_vector_type(8)))  float  v8f;

// ---- ordered-uint encoding for float atomicMax (handles negatives) ----
__device__ __forceinline__ unsigned f2ord(float f) {
  unsigned u = __float_as_uint(f);
  return (u & 0x80000000u) ? ~u : (u | 0x80000000u);
}
__device__ __forceinline__ float ord2f(unsigned e) {
  return (e & 0x80000000u) ? __uint_as_float(e & 0x7fffffffu)
                           : __uint_as_float(~e);
}
__device__ __forceinline__ float lrelu(float v) { return v > 0.f ? v : 0.2f * v; }

// ============================================================================
// Kernel 1: fused  h = relu(x@w_in^T + b);  proj = h@w_proj^T;  skip = h@w_skip^T
//           + per-head scores via cross-lane reduction.
// Weights converted to bf16 ONCE per block into LDS (shared by 4 waves);
// B fragments are single aligned 32B LDS loads; A fragments hoisted.
// ============================================================================
__global__ __launch_bounds__(128) void fused_gemm_kernel(
    const float* __restrict__ x, const float* __restrict__ w_in,
    const float* __restrict__ b_in, const float* __restrict__ w_proj,
    const float* __restrict__ a_src, const float* __restrict__ a_trg,
    const float* __restrict__ w_skip,
    float* __restrict__ proj_g, float* __restrict__ out_acc,
    float* __restrict__ ssrc_g, float* __restrict__ strg_g, int n_rows)
{
  __shared__ __bf16 wi[HSZ][WI_PAD];      // 69.6 KB  w_in  (bf16, padded rows)
  __shared__ __bf16 wp[OUTSZ][WH_PAD];    // 36.9 KB  w_proj
  __shared__ __bf16 wk[OUTSZ][WH_PAD];    // 36.9 KB  w_skip
  __shared__ __bf16 hb[4][16][WH_PAD];    // 18.4 KB  per-wave h tile (bf16)

  const int wave = threadIdx.x >> 5;
  const int lane = threadIdx.x & 31;
  const int mn   = lane & 15;   // row (A) / col (B,C,D) within 16-tile
  const int half = lane >> 4;   // half-wave selector
  const int row0 = (blockIdx.x * 4 + wave) * 16;
  const int srow = min(row0 + mn, n_rows - 1);   // clamped load row (EXEC stays full)

  // ---------- cooperative one-shot bf16 staging of all weights ----------
  for (int i = threadIdx.x * 4; i < HSZ * INSZ; i += blockDim.x * 4) {
    const int r = i >> 8, c = i & (INSZ - 1);
    const float4 w = *(const float4*)(w_in + (size_t)r * INSZ + c);
    __bf16* d = &wi[r][c];
    d[0] = (__bf16)w.x; d[1] = (__bf16)w.y; d[2] = (__bf16)w.z; d[3] = (__bf16)w.w;
  }
  for (int i = threadIdx.x * 4; i < OUTSZ * HSZ; i += blockDim.x * 4) {
    const int r = i >> 7, c = i & (HSZ - 1);
    const float4 p = *(const float4*)(w_proj + (size_t)r * HSZ + c);
    const float4 s = *(const float4*)(w_skip + (size_t)r * HSZ + c);
    __bf16* dp = &wp[r][c];
    __bf16* ds = &wk[r][c];
    dp[0] = (__bf16)p.x; dp[1] = (__bf16)p.y; dp[2] = (__bf16)p.z; dp[3] = (__bf16)p.w;
    ds[0] = (__bf16)s.x; ds[1] = (__bf16)s.y; ds[2] = (__bf16)s.z; ds[3] = (__bf16)s.w;
  }
  __syncthreads();

  // ---------------- GEMM 1: h[16x128] = relu(x_tile @ w_in^T + b) ----------------
  // A fragments depend only on k-step: load/convert once (8 x v16bf = 64 VGPRs).
  v16bf afrag[INSZ / 32];
  {
    const float* xrow = x + (size_t)srow * INSZ;
#pragma unroll
    for (int ks = 0; ks < INSZ / 32; ++ks) {
      const float* xr = xrow + ks * 32 + half * 16;
#pragma unroll
      for (int j = 0; j < 8; ++j) {
        afrag[ks][j]     = (__bf16)xr[j];
        afrag[ks][j + 8] = (__bf16)xr[8 + j];
      }
    }
  }
  for (int nt = 0; nt < HSZ / 16; ++nt) {
    v8f acc = {};
#pragma unroll
    for (int ks = 0; ks < INSZ / 32; ++ks) {
      const v16bf b = *(const v16bf*)&wi[nt * 16 + mn][ks * 32 + half * 16];
      acc = __builtin_amdgcn_wmma_f32_16x16x32_bf16(false, afrag[ks], false, b,
                                                    (short)0, acc, false, false);
    }
    const float bias = b_in[nt * 16 + mn];
#pragma unroll
    for (int r = 0; r < 8; ++r) {
      float v = acc[r] + bias;
      hb[wave][r + 8 * half][nt * 16 + mn] = (__bf16)(v > 0.f ? v : 0.f);  // ReLU
    }
  }
  __syncthreads();

  // ---------------- GEMM 2+3: proj / skip from bf16 h tile ----------------
  // A fragments: direct 16B LDS loads, no conversion (4 x v16bf = 32 VGPRs).
  v16bf a2[HSZ / 32];
#pragma unroll
  for (int ks = 0; ks < HSZ / 32; ++ks) {
    const __bf16* hr = &hb[wave][mn][ks * 32 + half * 16];
    const v8bf lo = *(const v8bf*)hr;
    const v8bf hi = *(const v8bf*)(hr + 8);
#pragma unroll
    for (int j = 0; j < 8; ++j) { a2[ks][j] = lo[j]; a2[ks][j + 8] = hi[j]; }
  }
  for (int nt = 0; nt < OUTSZ / 16; ++nt) {       // nt == head (tile width == HDIM)
    v8f accP = {}, accS = {};
#pragma unroll
    for (int ks = 0; ks < HSZ / 32; ++ks) {
      const v16bf bp = *(const v16bf*)&wp[nt * 16 + mn][ks * 32 + half * 16];
      const v16bf bs = *(const v16bf*)&wk[nt * 16 + mn][ks * 32 + half * 16];
      accP = __builtin_amdgcn_wmma_f32_16x16x32_bf16(false, a2[ks], false, bp,
                                                     (short)0, accP, false, false);
      accS = __builtin_amdgcn_wmma_f32_16x16x32_bf16(false, a2[ks], false, bs,
                                                     (short)0, accS, false, false);
    }
    // stores of proj / skip-initialized accumulator
#pragma unroll
    for (int r = 0; r < 8; ++r) {
      const int grow = row0 + r + 8 * half;
      const int col  = nt * 16 + mn;
      if (grow < n_rows) {
        proj_g[(size_t)grow * OUTSZ + col]  = accP[r];
        out_acc[(size_t)grow * OUTSZ + col] = accS[r];
      }
    }
    // per-head scores: s[m][nt] = sum_d proj[m][nt*16+d] * a[nt*16+d]
    // d == mn here, so butterfly-reduce across the 16 lanes of each half-wave.
    const float ca = a_src[nt * HDIM + mn];
    const float cb = a_trg[nt * HDIM + mn];
#pragma unroll
    for (int r = 0; r < 8; ++r) {
      float ss = accP[r] * ca;
      float st = accP[r] * cb;
#pragma unroll
      for (int m = 1; m < 16; m <<= 1) {
        ss += __shfl_xor(ss, m);
        st += __shfl_xor(st, m);
      }
      const int grow = row0 + r + 8 * half;
      if (mn == 0 && grow < n_rows) {
        ssrc_g[grow * NHEAD + nt] = ss;
        strg_g[grow * NHEAD + nt] = st;
      }
    }
  }
}

// ============================================================================
// Kernel 2: zero denominators, init global max
// ============================================================================
__global__ void init_kernel(float* __restrict__ denom, unsigned* __restrict__ gmax, int n) {
  const int i = blockIdx.x * blockDim.x + threadIdx.x;
  if (i < n) denom[i] = 0.f;
  if (i == 0) *gmax = f2ord(-__builtin_inff());
}

// ============================================================================
// Kernel 3: global max of leaky_relu(s_src[src]+s_trg[trg]) over all edges/heads
// ============================================================================
__global__ __launch_bounds__(256) void edge_max_kernel(
    const int* __restrict__ adj, const float* __restrict__ ssrc,
    const float* __restrict__ strg, unsigned* __restrict__ gmax, int E)
{
  const int e = blockIdx.x * blockDim.x + threadIdx.x;
  float m = -__builtin_inff();
  if (e < E) {
    const int src = adj[e], trg = adj[E + e];
#pragma unroll
    for (int h = 0; h < NHEAD; ++h)
      m = fmaxf(m, lrelu(ssrc[src * NHEAD + h] + strg[trg * NHEAD + h]));
  }
#pragma unroll
  for (int off = 16; off > 0; off >>= 1) m = fmaxf(m, __shfl_down(m, off));
  if ((threadIdx.x & 31) == 0) atomicMax(gmax, f2ord(m));
}

// ============================================================================
// Kernel 4: denom[trg,h] += exp(score - gmax)
// ============================================================================
__global__ __launch_bounds__(256) void edge_denom_kernel(
    const int* __restrict__ adj, const float* __restrict__ ssrc,
    const float* __restrict__ strg, const unsigned* __restrict__ gmax,
    float* __restrict__ denom, int E)
{
  const int e = blockIdx.x * blockDim.x + threadIdx.x;
  if (e >= E) return;
  const float mx = ord2f(*gmax);
  const int src = adj[e], trg = adj[E + e];
#pragma unroll
  for (int h = 0; h < NHEAD; ++h) {
    const float sc = lrelu(ssrc[src * NHEAD + h] + strg[trg * NHEAD + h]);
    atomicAdd(&denom[trg * NHEAD + h], __expf(sc - mx));
  }
}

// ============================================================================
// Kernel 5: out_acc[trg] += proj[src] * att   (one wave per edge, float4/lane;
//           lane>>2 = head, so each lane recomputes its head's attention)
// ============================================================================
__global__ __launch_bounds__(256) void edge_aggr_kernel(
    const int* __restrict__ adj, const float* __restrict__ ssrc,
    const float* __restrict__ strg, const unsigned* __restrict__ gmax,
    const float* __restrict__ denom, const float* __restrict__ proj,
    float* __restrict__ out_acc, int E)
{
  const int wid  = (int)((blockIdx.x * blockDim.x + threadIdx.x) >> 5);
  const int lane = threadIdx.x & 31;
  if (wid >= E) return;
  const int src = adj[wid], trg = adj[E + wid];
  const int head = lane >> 2;                       // 4 lanes per head (4 floats each)
  const float mx = ord2f(*gmax);
  const float sc = lrelu(ssrc[src * NHEAD + head] + strg[trg * NHEAD + head]);
  const float att = __expf(sc - mx) / (denom[trg * NHEAD + head] + 1e-16f);
  const float4 p = *(const float4*)(proj + (size_t)src * OUTSZ + lane * 4);
  float* o = out_acc + (size_t)trg * OUTSZ + lane * 4;
  atomicAdd(o + 0, p.x * att);
  atomicAdd(o + 1, p.y * att);
  atomicAdd(o + 2, p.z * att);
  atomicAdd(o + 3, p.w * att);
}

// ============================================================================
// Kernel 6: ELU + index_select
// ============================================================================
__global__ __launch_bounds__(256) void finalize_kernel(
    const int* __restrict__ index0, const float* __restrict__ out_acc,
    float* __restrict__ out, int total)
{
  const int i = blockIdx.x * blockDim.x + threadIdx.x;
  if (i >= total) return;
  const int r = i >> 7, c = i & (OUTSZ - 1);
  const float v = out_acc[(size_t)index0[r] * OUTSZ + c];
  out[i] = v > 0.f ? v : (__expf(v) - 1.f);
}

// ============================================================================
extern "C" void kernel_launch(void* const* d_in, const int* in_sizes, int n_in,
                              void* d_out, int out_size, void* d_ws, size_t ws_size,
                              hipStream_t stream) {
  const float* x      = (const float*)d_in[0];
  const int*   adj    = (const int*)  d_in[1];
  const int*   index0 = (const int*)  d_in[2];
  const float* w_in   = (const float*)d_in[3];
  const float* b_in   = (const float*)d_in[4];
  const float* w_proj = (const float*)d_in[5];
  const float* a_src  = (const float*)d_in[6];
  const float* a_trg  = (const float*)d_in[7];
  const float* w_skip = (const float*)d_in[8];
  float* out = (float*)d_out;

  const int N = in_sizes[0] / INSZ;
  const int E = in_sizes[1] / 2;
  const int K = in_sizes[2];

  // workspace carve-up (~56 MB)
  float* proj    = (float*)d_ws;                       // N * 128
  float* out_acc = proj    + (size_t)N * OUTSZ;        // N * 128
  float* ssrc    = out_acc + (size_t)N * OUTSZ;        // N * 8
  float* strg    = ssrc    + (size_t)N * NHEAD;        // N * 8
  float* denom   = strg    + (size_t)N * NHEAD;        // N * 8
  unsigned* gmax = (unsigned*)(denom + (size_t)N * NHEAD);

  const int gemm_blocks = (N + 63) / 64;               // 64 rows per block (4 waves)
  fused_gemm_kernel<<<gemm_blocks, 128, 0, stream>>>(
      x, w_in, b_in, w_proj, a_src, a_trg, w_skip, proj, out_acc, ssrc, strg, N);

  const int nden = N * NHEAD;
  init_kernel<<<(nden + 255) / 256, 256, 0, stream>>>(denom, gmax, nden);

  edge_max_kernel  <<<(E + 255) / 256, 256, 0, stream>>>(adj, ssrc, strg, gmax, E);
  edge_denom_kernel<<<(E + 255) / 256, 256, 0, stream>>>(adj, ssrc, strg, gmax, denom, E);
  edge_aggr_kernel <<<(E + 7) / 8,     256, 0, stream>>>(adj, ssrc, strg, gmax, denom,
                                                         proj, out_acc, E);
  const int total = K * OUTSZ;
  finalize_kernel<<<(total + 255) / 256, 256, 0, stream>>>(index0, out_acc, out, total);
}
